// Model_9328668967273
// MI455X (gfx1250) — matrix operand order repaired and checked by reading, not compiled, not run on hardware
//
#include <hip/hip_runtime.h>
#include <stdint.h>

// ---------------------------------------------------------------------------
// JK flip-flop RNN, T=2048 sequential steps.  Weights (63.5 MB bf16) are
// L2-resident on MI455X (192 MB L2), so a single persistent kernel loops over
// time with grid-wide barriers and streams weights from L2 each step.
// Mat-vecs run on v_wmma_f32_16x16x32_bf16 with the activation vector
// broadcast across all 16 A rows; weights are pre-packed into the WMMA-B
// register layout so each lane loads 32 contiguous bytes per K-chunk.
// Assumes ws_size >= ~64 MB (weight cache + tiny state).
// ---------------------------------------------------------------------------

#define T_STEPS  2048
#define IN_SZ    512
#define H_SZ     2048
#define KZ       2560   // [h2 ; x_t]        -> W1 rows
#define KC1      3072   // [z ; out ; h1]    -> J1/K1 rows
#define KC2      4096   // [h1' ; h2]        -> J2/K2 rows
#define NBLOCKS  64
#define NTHREADS 128    // 4 waves (wave32)

typedef __attribute__((ext_vector_type(16))) __bf16       v16bf;
typedef __attribute__((ext_vector_type(8)))  float        v8f;
typedef __attribute__((ext_vector_type(4)))  unsigned int v4u;
typedef __attribute__((ext_vector_type(8)))  unsigned int v8u;

__device__ __forceinline__ unsigned short f2bf(float f) {
  unsigned u = __builtin_bit_cast(unsigned, f);
  if ((u & 0x7F800000u) != 0x7F800000u)          // not inf/nan: round-to-nearest-even
    u += 0x7FFFu + ((u >> 16) & 1u);
  return (unsigned short)(u >> 16);
}

__device__ __forceinline__ float sigmoidf_(float x) {
  return 1.0f / (1.0f + __expf(-x));
}

// ---------------------------------------------------------------------------
// Pack fp32 weight matrix (K x N, row-major) into bf16 WMMA-B layout:
//   dst[((nt*KC + kc)*32 + L)*16 + e] = bf16( W[(kc*32 + ((L>>4)<<4) + e)*N + nt*16 + (L&15)] )
// so lane L of a wave loads 32 contiguous bytes (one column, 16 K values).
// ---------------------------------------------------------------------------
__global__ void pack_b(const float* __restrict__ W, unsigned short* __restrict__ Wp,
                       int K, int N) {
  const int KC = K >> 5;
  const size_t total = (size_t)K * (size_t)N;
  for (size_t d = (size_t)blockIdx.x * blockDim.x + threadIdx.x; d < total;
       d += (size_t)gridDim.x * blockDim.x) {
    const int e  = (int)(d & 15u);
    const int L  = (int)((d >> 4) & 31u);
    const size_t blk = d >> 9;
    const int kc = (int)(blk % (size_t)KC);
    const int nt = (int)(blk / (size_t)KC);
    const int n = nt * 16 + (L & 15);
    const int k = kc * 32 + ((L >> 4) << 4) + e;
    Wp[d] = f2bf(W[(size_t)k * N + n]);
  }
}

__global__ void init_state(unsigned short* c1b, unsigned short* c2b,
                           unsigned short* zxb, unsigned short* h2b,
                           float* h1f, float* h2f,
                           const float* __restrict__ X, unsigned* sync_) {
  const int i = blockIdx.x * blockDim.x + threadIdx.x;   // 0..4095
  if (i < 2)     sync_[i] = 0u;
  if (i < H_SZ)  { h1f[i] = 0.f; h2f[i] = 0.f; h2b[i] = 0; zxb[i] = 0; }
  if (i < IN_SZ) zxb[H_SZ + i] = f2bf(X[i]);             // x_0
  if (i < KC1)   c1b[i] = 0;
  if (i < KC2)   c2b[i] = 0;
}

// ---------------------------------------------------------------------------
// Device-wide sense-reversing barrier (64 resident workgroups).
// ---------------------------------------------------------------------------
__device__ __forceinline__ void grid_barrier(unsigned* sync_, unsigned nb) {
  __threadfence();
  __syncthreads();
  if (threadIdx.x == 0) {
    const unsigned g = __atomic_load_n(sync_ + 1, __ATOMIC_ACQUIRE);
    if (__atomic_fetch_add(sync_ + 0, 1u, __ATOMIC_ACQ_REL) == nb - 1u) {
      __atomic_store_n(sync_ + 0, 0u, __ATOMIC_RELAXED);
      __atomic_fetch_add(sync_ + 1, 1u, __ATOMIC_ACQ_REL);
    } else {
      while (__atomic_load_n(sync_ + 1, __ATOMIC_ACQUIRE) == g)
        __builtin_amdgcn_s_sleep(2);
    }
  }
  __syncthreads();
  __threadfence();
}

__device__ __forceinline__ void lds_copy(unsigned short* dst,
                                         const unsigned short* __restrict__ src,
                                         int nhalves) {
  const int nc = nhalves >> 3;                 // 16B chunks
  for (int c = threadIdx.x; c < nc; c += blockDim.x)
    ((v4u*)dst)[c] = ((const v4u*)src)[c];
}

// ---------------------------------------------------------------------------
// One wave computes 16 outputs: out[ntile*16 + (lane&15)] = x . W[:,col].
// A: bf16 x chunk broadcast to all 16 rows (contiguous k = 16*(lane>>4) + e, same order as B); B: packed.
// ---------------------------------------------------------------------------
__device__ __forceinline__ float wave_matvec(const unsigned short* sx, int K,
                                             const unsigned short* __restrict__ Wp,
                                             int ntile, int lane) {
  v8f acc = {0.f, 0.f, 0.f, 0.f, 0.f, 0.f, 0.f, 0.f};
  const int kc_count = K >> 5;
  const unsigned short* bp =
      Wp + (((size_t)ntile * kc_count) * 32u + (unsigned)lane) * 16u;
  const int abase = (lane & 16) ? 16 : 0;
  for (int kc = 0; kc < kc_count; ++kc) {
    const unsigned short* ap = sx + (kc << 5) + abase;
    v4u a0 = *(const v4u*)(ap);
    v4u a1 = *(const v4u*)(ap + 8);
    v4u b0 = *(const v4u*)(bp);
    v4u b1 = *(const v4u*)(bp + 8);
    __builtin_prefetch(bp + 512 * 8, 0, 3);     // global_prefetch_b8, 8 chunks ahead
    v8u au = __builtin_shufflevector(a0, a1, 0, 1, 2, 3, 4, 5, 6, 7);
    v8u bu = __builtin_shufflevector(b0, b1, 0, 1, 2, 3, 4, 5, 6, 7);
    v16bf av = __builtin_bit_cast(v16bf, au);
    v16bf bv = __builtin_bit_cast(v16bf, bu);
    acc = __builtin_amdgcn_wmma_f32_16x16x32_bf16(false, av, false, bv,
                                                  (short)0, acc, false, false);
    bp += 512;                                  // next K-chunk (32 lanes * 16 halves)
  }
  return acc[0];                                // M-rows identical (broadcast A)
}

// ---------------------------------------------------------------------------
// Persistent time-loop kernel.  3 phases / step, grid barrier between phases.
// ---------------------------------------------------------------------------
__launch_bounds__(NTHREADS, 1)
__global__ void ffrnn_persistent(
    const unsigned short* __restrict__ W1p, const unsigned short* __restrict__ J1p,
    const unsigned short* __restrict__ K1p, const unsigned short* __restrict__ J2p,
    const unsigned short* __restrict__ K2p, const unsigned short* __restrict__ W2p,
    const float* __restrict__ b1, const float* __restrict__ j1bias,
    const float* __restrict__ k1bias, const float* __restrict__ j2bias,
    const float* __restrict__ k2bias, const float* __restrict__ b2,
    const float* __restrict__ X, float* __restrict__ Y,
    unsigned short* c1b, unsigned short* c2b, unsigned short* zxb,
    unsigned short* h2b, float* h1f, float* h2f, unsigned* sync_) {
  __shared__ __align__(16) unsigned short sx[KC2];   // staged activation vector
  __shared__ float sk[2][16];                        // K-gate exchange per tile pair

  const int bid  = blockIdx.x;
  const int wave = threadIdx.x >> 5;
  const int lane = threadIdx.x & 31;
  const int ln   = lane & 15;
  const unsigned nb = gridDim.x;

  for (int t = 0; t <= T_STEPS; ++t) {
    // ---- Phase 1: z = [h2;x_t]@W1+b1  and  out[t-1] = sigmoid(h2@W2+b2) ----
    if (bid < 16)
      lds_copy(sx, (bid < 8) ? zxb : h2b, (bid < 8) ? KZ : H_SZ);
    __syncthreads();
    if (bid < 8) {
      if (t < T_STEPS) {
        const int zt = bid * 4 + wave;               // 0..31 (512 outputs)
        const int n  = zt * 16 + ln;
        float v = wave_matvec(sx, KZ, W1p, zt, lane);
        if (lane < 16) c1b[n] = f2bf(v + b1[n]);
      }
    } else if (bid < 16) {
      if (t >= 1) {
        const int ot = (bid - 8) * 4 + wave;         // 0..31
        const int n  = ot * 16 + ln;
        float v = wave_matvec(sx, H_SZ, W2p, ot, lane);
        if (lane < 16) {
          float o = sigmoidf_(v + b2[n]);
          Y[(size_t)(t - 1) * IN_SZ + n] = o;
          c1b[IN_SZ + n] = f2bf(o);                  // out_prev for c1
        }
      }
    }
    if (t == T_STEPS) break;                         // epilogue wrote out[T-1]
    grid_barrier(sync_, nb);

    // ---- Phase 2: j1/k1 gates over c1 = [z;out;h1]; update h1 ----
    lds_copy(sx, c1b, KC1);
    __syncthreads();
    {
      const int gw  = bid * 4 + wave;                // 0..255
      const int tt  = gw >> 1;                       // 0..127 tiles
      const int isK = gw & 1;
      const int n   = tt * 16 + ln;
      float v = wave_matvec(sx, KC1, isK ? K1p : J1p, tt, lane);
      if (isK && lane < 16) sk[wave >> 1][ln] = sigmoidf_(v + k1bias[n]);
      __syncthreads();
      if (!isK && lane < 16) {
        float j  = sigmoidf_(v + j1bias[n]);
        float kk = sk[wave >> 1][ln];
        float h  = h1f[n];
        float hn = j * (1.0f - h) + (1.0f - kk) * h;
        h1f[n] = hn;
        c2b[n] = f2bf(hn);                           // c2 = [h1' ; h2_old]
        c2b[H_SZ + n] = h2b[n];                      // deferred h2_old copy
      }
    }
    grid_barrier(sync_, nb);

    // ---- Phase 3: j2/k2 gates over c2; update h2; stage next-step vectors ----
    lds_copy(sx, c2b, KC2);
    __syncthreads();
    {
      const int gw  = bid * 4 + wave;
      const int tt  = gw >> 1;
      const int isK = gw & 1;
      const int n   = tt * 16 + ln;
      float v = wave_matvec(sx, KC2, isK ? K2p : J2p, tt, lane);
      if (isK && lane < 16) sk[wave >> 1][ln] = sigmoidf_(v + k2bias[n]);
      __syncthreads();
      if (!isK && lane < 16) {
        float j  = sigmoidf_(v + j2bias[n]);
        float kk = sk[wave >> 1][ln];
        float h  = h2f[n];
        float hn = j * (1.0f - h) + (1.0f - kk) * h;
        h2f[n] = hn;
        const unsigned short hb = f2bf(hn);
        h2b[n] = hb;                                 // W2 input next P1
        zxb[n] = hb;                                 // W1 input next P1
        c1b[2 * IN_SZ + n] = f2bf(h1f[n]);           // h1 section of next c1
        if (n < IN_SZ && (t + 1) < T_STEPS)
          zxb[H_SZ + n] = f2bf(X[(size_t)(t + 1) * IN_SZ + n]);
      }
    }
    grid_barrier(sync_, nb);
  }
}

extern "C" void kernel_launch(void* const* d_in, const int* in_sizes, int n_in,
                              void* d_out, int out_size, void* d_ws, size_t ws_size,
                              hipStream_t stream) {
  (void)in_sizes; (void)n_in; (void)out_size; (void)ws_size;
  const float* X   = (const float*)d_in[0];
  const float* W1  = (const float*)d_in[1];
  const float* b1  = (const float*)d_in[2];
  const float* J1W = (const float*)d_in[3];
  const float* J1b = (const float*)d_in[4];
  const float* K1W = (const float*)d_in[5];
  const float* K1b = (const float*)d_in[6];
  const float* J2W = (const float*)d_in[7];
  const float* J2b = (const float*)d_in[8];
  const float* K2W = (const float*)d_in[9];
  const float* K2b = (const float*)d_in[10];
  const float* W2  = (const float*)d_in[11];
  const float* b2  = (const float*)d_in[12];
  float* Y = (float*)d_out;

  char* p = (char*)d_ws;
  size_t off = 0;
  auto take = [&](size_t bytes) {
    char* r = p + off;
    off = (off + bytes + 255) & ~(size_t)255;
    return r;
  };
  unsigned short* W1p = (unsigned short*)take((size_t)KZ  * IN_SZ * 2);
  unsigned short* J1p = (unsigned short*)take((size_t)KC1 * H_SZ  * 2);
  unsigned short* K1p = (unsigned short*)take((size_t)KC1 * H_SZ  * 2);
  unsigned short* J2p = (unsigned short*)take((size_t)KC2 * H_SZ  * 2);
  unsigned short* K2p = (unsigned short*)take((size_t)KC2 * H_SZ  * 2);
  unsigned short* W2p = (unsigned short*)take((size_t)H_SZ * IN_SZ * 2);
  unsigned short* c1b = (unsigned short*)take(KC1 * 2);
  unsigned short* c2b = (unsigned short*)take(KC2 * 2);
  unsigned short* zxb = (unsigned short*)take(KZ * 2);
  unsigned short* h2b = (unsigned short*)take(H_SZ * 2);
  float* h1f = (float*)take(H_SZ * 4);
  float* h2f = (float*)take(H_SZ * 4);
  unsigned* syncp = (unsigned*)take(256);

  pack_b<<<2048, 256, 0, stream>>>(W1,  W1p, KZ,  IN_SZ);
  pack_b<<<2048, 256, 0, stream>>>(J1W, J1p, KC1, H_SZ);
  pack_b<<<2048, 256, 0, stream>>>(K1W, K1p, KC1, H_SZ);
  pack_b<<<2048, 256, 0, stream>>>(J2W, J2p, KC2, H_SZ);
  pack_b<<<2048, 256, 0, stream>>>(K2W, K2p, KC2, H_SZ);
  pack_b<<<2048, 256, 0, stream>>>(W2,  W2p, H_SZ, IN_SZ);
  init_state<<<16, 256, 0, stream>>>(c1b, c2b, zxb, h2b, h1f, h2f, X, syncp);

  ffrnn_persistent<<<NBLOCKS, NTHREADS, 0, stream>>>(
      W1p, J1p, K1p, J2p, K2p, W2p,
      b1, J1b, K1b, J2b, K2b, b2,
      X, Y, c1b, c2b, zxb, h2b, h1f, h2f, syncp);
}
